// SwinBlock_91044716740964
// MI455X (gfx1250) — matrix operand order repaired and checked by reading, not compiled, not run on hardware
//
#include <hip/hip_runtime.h>

typedef __bf16 bf16;
typedef __attribute__((ext_vector_type(16))) __bf16 v16bf;
typedef __attribute__((ext_vector_type(8)))  __bf16 v8bf;
typedef __attribute__((ext_vector_type(8)))  float  v8f;

#define WMMA_BF16(a,b,c) __builtin_amdgcn_wmma_f32_16x16x32_bf16(false,(a),false,(b),(short)0,(c),false,false)

#define T_TOK 65536
#define SHUF16 0,1,2,3,4,5,6,7,8,9,10,11,12,13,14,15

// ---------------------------------------------------------------------------
// Fragment loaders (CDNA5 WMMA wave32 layouts, 16-bit operands)
// A 16x32: lane m = lane&15, kb=(lane>>4)*8; elems 0..7 -> K=kb+e, 8..15 -> K=kb+8+e
// B 32x16: lane n = lane&15, kg=lane>>4;     elems 0..7 -> K=kg*8+e, 8..15 -> K=kg*8+8+e
// D 16x16 f32: elem r -> M = r + 8*(lane>>4), N = lane&15
// ---------------------------------------------------------------------------
__device__ __forceinline__ v16bf ld_a_frag(const bf16* base, int ld, int row0, int lane) {
    int m  = row0 + (lane & 15);
    int kb = (lane >> 4) * 8;
    const bf16* p = base + m * ld + kb;
    v8bf x0 = *(const v8bf*)(p);
    v8bf x1 = *(const v8bf*)(p + 16);
    return __builtin_shufflevector(x0, x1, SHUF16);
}

__device__ __forceinline__ v16bf ld_b_frag(const bf16* base, int ld, int col0, int lane) {
    int n  = col0 + (lane & 15);
    int kb = (lane >> 4) * 8;
    const bf16* p = base + n * ld + kb;
    v8bf x0 = *(const v8bf*)(p);
    v8bf x1 = *(const v8bf*)(p + 16);
    return __builtin_shufflevector(x0, x1, SHUF16);
}

// issue one 16B async global->LDS copy (per lane), tracked by ASYNCcnt
__device__ __forceinline__ void async_copy_b128(const bf16* gptr, bf16* lptr) {
    unsigned lds_off = (unsigned)(unsigned long long)lptr;       // flat LDS aperture: low 32 bits
    unsigned long long ga = (unsigned long long)gptr;
    asm volatile("global_load_async_to_lds_b128 %0, %1, off"
                 :: "v"(lds_off), "v"(ga) : "memory");
}

__device__ __forceinline__ void wait_async0() {
    asm volatile("s_wait_asynccnt 0x0" ::: "memory");
}

// ---------------------------------------------------------------------------
// Generic token-major GEMM: Out[t][n] = act( sum_k A[t][k] * W[n][k] + bias[n] )
// A: bf16 [T][K], W: bf16 [N][K], Out: f32 or bf16 [T][N]
// block = 256 threads (8 waves), tile 128(M) x 64(N), K step 32,
// double-buffered LDS fed by async global->LDS copies (one barrier / K step)
// ---------------------------------------------------------------------------
template<bool RELU, bool OBF16>
__global__ __launch_bounds__(256) void gemm_wmma_kernel(
        const bf16* __restrict__ A, const bf16* __restrict__ W,
        const float* __restrict__ bias, void* __restrict__ Out,
        int N, int K) {
    __shared__ bf16 As[2][128 * 40];
    __shared__ bf16 Ws[2][64 * 40];

    const int tid  = threadIdx.x;
    const int lane = tid & 31;
    const int wv   = tid >> 5;
    const int wm   = (wv & 3) * 32;   // wave row offset in tile
    const int wn   = (wv >> 2) * 32;  // wave col offset in tile
    const int t0   = blockIdx.x * 128;
    const int n0   = blockIdx.y * 64;

    const int row = tid >> 2, seg = tid & 3;

    auto stage = [&](int k0, int buf) {
        // A tile: 128 x 32 bf16 = 2 x (64 rows x 4 x 16B); W tile: 64 x 32
        async_copy_b128(A + (size_t)(t0 + row) * K + k0 + seg * 8,
                        &As[buf][row * 40 + seg * 8]);
        async_copy_b128(A + (size_t)(t0 + row + 64) * K + k0 + seg * 8,
                        &As[buf][(row + 64) * 40 + seg * 8]);
        async_copy_b128(W + (size_t)(n0 + row) * K + k0 + seg * 8,
                        &Ws[buf][row * 40 + seg * 8]);
    };

    v8f acc00 = {}, acc01 = {}, acc10 = {}, acc11 = {};

    const int nsteps = K >> 5;
    stage(0, 0);
    for (int s = 0; s < nsteps; ++s) {
        wait_async0();        // my copies for buffer s&1 have landed
        __syncthreads();      // everyone's copies landed; everyone done reading buf (s&1) from step s-2
        if (s + 1 < nsteps) stage((s + 1) << 5, (s + 1) & 1);  // overlaps with compute below

        const bf16* Ab = As[s & 1];
        const bf16* Wb = Ws[s & 1];
        v16bf a0 = ld_a_frag(Ab, 40, wm,      lane);
        v16bf a1 = ld_a_frag(Ab, 40, wm + 16, lane);
        v16bf b0 = ld_b_frag(Wb, 40, wn,      lane);
        v16bf b1 = ld_b_frag(Wb, 40, wn + 16, lane);

        acc00 = WMMA_BF16(a0, b0, acc00);
        acc01 = WMMA_BF16(a0, b1, acc01);
        acc10 = WMMA_BF16(a1, b0, acc10);
        acc11 = WMMA_BF16(a1, b1, acc11);
    }

    const int hi = lane >> 4, nl = lane & 15;
#pragma unroll
    for (int i = 0; i < 2; ++i) {
#pragma unroll
        for (int j = 0; j < 2; ++j) {
            const v8f& c = (i == 0) ? (j == 0 ? acc00 : acc01)
                                    : (j == 0 ? acc10 : acc11);
#pragma unroll
            for (int r = 0; r < 8; ++r) {
                int t = t0 + wm + i * 16 + r + hi * 8;
                int n = n0 + wn + j * 16 + nl;
                float v = c[r] + bias[n];
                if (RELU) v = fmaxf(v, 0.f);
                if (OBF16) ((bf16*)Out)[(size_t)t * N + n] = (bf16)v;
                else       ((float*)Out)[(size_t)t * N + n] = v;
            }
        }
    }
}

// ---------------------------------------------------------------------------
// Positional + time embeddings, token-major bf16 [T][512]
// ---------------------------------------------------------------------------
__global__ __launch_bounds__(256) void embs_kernel(const float* __restrict__ tvec,
                                                   bf16* __restrict__ embs) {
    int gid = blockIdx.x * 256 + threadIdx.x;  // T*512 total
    int c = gid & 511;
    int t = gid >> 9;
    int b = t >> 14;
    int hw = t & 16383;
    int h = hw >> 7, w = hw & 127;
    float val;
    if (c < 256) {
        int pos = (c < 128) ? h : w;
        int cc  = c & 127;
        int i   = cc & 63;
        float freq = __expf(-logf(10000.f) * (2.f * (float)i / 128.f));
        float ang  = (float)pos * freq;
        val = (cc >= 64) ? __cosf(ang) : __sinf(ang);
    } else {
        int i2 = c - 256;
        int ii = i2 & 127;
        float freq = __expf(-logf(10000.f) * ((float)ii / 128.f));
        float ang  = tvec[b] * freq;
        val = (i2 >= 128) ? __cosf(ang) : __sinf(ang);
    }
    embs[gid] = (bf16)val;
}

// ---------------------------------------------------------------------------
// x = x*mul + bias; res = x; xn = (x - mean)/sqrt(var_ddof1 + eps) * scale
// Processes 32 tokens per block via LDS transpose tile.
// x: (B,C,H,W) f32, mb: [T][512] f32 (mul|bias), res: [T][256] f32, xn: [T][256] bf16
// ---------------------------------------------------------------------------
__global__ __launch_bounds__(256) void affine_ln_kernel(
        const float* __restrict__ x, const float* __restrict__ mb,
        const float* __restrict__ scale, float* __restrict__ res,
        bf16* __restrict__ xn) {
    __shared__ float xs[256 * 33];
    __shared__ float red[2][8][32];
    __shared__ float stat[2][32];

    const int tid = threadIdx.x;
    const int t0  = blockIdx.x * 32;
    const int b   = t0 >> 14;
    const int hw0 = t0 & 16383;

    // load x tile: 8 channels x 32 tokens per iter (coalesced along hw)
    for (int it = 0; it < 32; ++it) {
        int idx = it * 256 + tid;
        int c = idx >> 5, tk = idx & 31;
        xs[c * 33 + tk] = x[((size_t)(b * 256 + c)) * 16384 + hw0 + tk];
    }
    __syncthreads();
    // apply mul/bias (coalesced along channel of mb row)
    for (int j = 0; j < 32; ++j) {
        int idx = j * 256 + tid;
        int tk = idx >> 8, c = idx & 255;
        size_t rowb = (size_t)(t0 + tk) * 512;
        float mu  = mb[rowb + c];
        float bia = mb[rowb + 256 + c];
        xs[c * 33 + tk] = xs[c * 33 + tk] * mu + bia;
    }
    __syncthreads();
    // partial reductions: 8 parts x 32 tokens
    {
        int tk = tid & 31, part = tid >> 5;
        float s = 0.f, ss = 0.f;
        for (int c = part * 32; c < part * 32 + 32; ++c) {
            float v = xs[c * 33 + tk];
            s += v; ss += v * v;
        }
        red[0][part][tk] = s; red[1][part][tk] = ss;
    }
    __syncthreads();
    if (tid < 32) {
        float s = 0.f, ss = 0.f;
        for (int p = 0; p < 8; ++p) { s += red[0][p][tid]; ss += red[1][p][tid]; }
        float mean = s * (1.f / 256.f);
        float var  = (ss - 256.f * mean * mean) * (1.f / 255.f);
        stat[0][tid] = mean;
        stat[1][tid] = rsqrtf(var + 1e-4f);
    }
    __syncthreads();
    for (int j = 0; j < 32; ++j) {
        int idx = j * 256 + tid;
        int tk = idx >> 8, c = idx & 255;
        float y = xs[c * 33 + tk];
        size_t o = (size_t)(t0 + tk) * 256 + c;
        res[o] = y;
        xn[o] = (bf16)((y - stat[0][tk]) * stat[1][tk] * scale[c]);
    }
}

// ---------------------------------------------------------------------------
// Shifted-window attention, one wave per (window, head).
// qkv: bf16 [T][768] (q|k|v, each [heads=8][hd=32]); out: bf16 [T][256]
// ---------------------------------------------------------------------------
__global__ __launch_bounds__(256) void attn_kernel(const bf16* __restrict__ qkv,
                                                   const float* __restrict__ rel_bias,
                                                   bf16* __restrict__ outp) {
    __shared__ bf16 P[8][16 * 24];

    const int tid = threadIdx.x, lane = tid & 31, wv = tid >> 5;
    const int gid  = blockIdx.x * 8 + wv;
    const int head = gid & 7;
    const int win  = gid >> 3;
    const int b    = win >> 10;
    const int wi   = win & 1023;
    const int wy   = wi >> 5, wx = wi & 31;
    const int hi   = lane >> 4, nl = lane & 15;

    auto tok = [&](int i) -> int {
        int hh = (wy * 4 + (i >> 2) + 2) & 127;   // SHIFT=2, roll(-2)
        int ww = (wx * 4 + (i & 3) + 2) & 127;
        return (b << 14) + (hh << 7) + ww;
    };

    // Q fragment (A: 16 tokens x 32 dims)
    v16bf aq;
    {
        const bf16* p = qkv + (size_t)tok(nl) * 768 + head * 32 + hi * 8;
        v8bf x0 = *(const v8bf*)p;
        v8bf x1 = *(const v8bf*)(p + 16);
        aq = __builtin_shufflevector(x0, x1, SHUF16);
    }
    // K^T fragment (B: 32 dims x 16 tokens)
    v16bf bk;
    {
        const bf16* p = qkv + (size_t)tok(nl) * 768 + 256 + head * 32 + hi * 8;
        v8bf x0 = *(const v8bf*)p;
        v8bf x1 = *(const v8bf*)(p + 16);
        bk = __builtin_shufflevector(x0, x1, SHUF16);
    }
    v8f zc = {};
    v8f s = WMMA_BF16(aq, bk, zc);

    // scale + relative position bias + softmax over key tokens (16 lanes/half)
    float pr[8];
#pragma unroll
    for (int r = 0; r < 8; ++r) {
        int m = r + hi * 8;
        int dy = (m >> 2) - (nl >> 2) + 3;
        int dx = (m & 3) - (nl & 3) + 3;
        float val = s[r] * 0.17677669529663687f + rel_bias[head * 49 + dy * 7 + dx];
        float mx = val;
        for (int msk = 1; msk < 16; msk <<= 1) mx = fmaxf(mx, __shfl_xor(mx, msk, 32));
        float e = __expf(val - mx);
        float sm = e;
        for (int msk = 1; msk < 16; msk <<= 1) sm += __shfl_xor(sm, msk, 32);
        pr[r] = e / sm;
    }
    // redistribute P into A-fragment layout via LDS
#pragma unroll
    for (int r = 0; r < 8; ++r) {
        int m = r + hi * 8;
        P[wv][m * 24 + nl] = (bf16)pr[r];
    }
    __syncthreads();
    v16bf ap;
    {
        const bf16* p = &P[wv][nl * 24 + hi * 8];
        v8bf x0 = *(const v8bf*)p;
        v8bf z;
#pragma unroll
        for (int e = 0; e < 8; ++e) z[e] = (bf16)0.f;
        ap = __builtin_shufflevector(x0, z, SHUF16);  // K=16..31 zero padded
    }
    __syncthreads();

    // O = P x V  (two 16-wide d tiles, K padded to 32)
#pragma unroll
    for (int j = 0; j < 2; ++j) {
        int dg = head * 32 + j * 16 + nl;
        v16bf bv;
#pragma unroll
        for (int e = 0; e < 16; ++e)
            bv[e] = (e < 8) ? qkv[(size_t)tok(hi * 8 + e) * 768 + 512 + dg] : (bf16)0.f;
        v8f zc2 = {};
        v8f o = WMMA_BF16(ap, bv, zc2);
#pragma unroll
        for (int r = 0; r < 8; ++r) {
            int m = r + hi * 8;
            outp[(size_t)tok(m) * 256 + dg] = (bf16)o[r];
        }
    }
}

// ---------------------------------------------------------------------------
// fa = fa * fb  (fb already ReLU'd by GEMM epilogue)
// ---------------------------------------------------------------------------
__global__ __launch_bounds__(256) void gate_mul_kernel(bf16* __restrict__ fa,
                                                       const bf16* __restrict__ fb) {
    size_t i = (size_t)blockIdx.x * 256 + threadIdx.x;
    fa[i] = (bf16)((float)fa[i] * (float)fb[i]);
}

// ---------------------------------------------------------------------------
// out(B,C,H,W) = attn + ffn + res (token-major inputs), LDS transpose tile
// ---------------------------------------------------------------------------
__global__ __launch_bounds__(256) void final_add_kernel(
        const float* __restrict__ attnf, const float* __restrict__ ffnf,
        const float* __restrict__ res, float* __restrict__ out) {
    __shared__ float s[256 * 33];
    const int tid = threadIdx.x;
    const int t0  = blockIdx.x * 32;
    const int b   = t0 >> 14;
    const int hw0 = t0 & 16383;

    for (int j = 0; j < 32; ++j) {
        int idx = j * 256 + tid;
        int tk = idx >> 8, c = idx & 255;
        size_t o = (size_t)(t0 + tk) * 256 + c;
        s[c * 33 + tk] = attnf[o] + ffnf[o] + res[o];
    }
    __syncthreads();
    for (int it = 0; it < 32; ++it) {
        int idx = it * 256 + tid;
        int c = idx >> 5, tk = idx & 31;
        out[((size_t)(b * 256 + c)) * 16384 + hw0 + tk] = s[c * 33 + tk];
    }
}

// ---------------------------------------------------------------------------
// f32 -> bf16 weight conversion
// ---------------------------------------------------------------------------
__global__ __launch_bounds__(256) void f2bf_kernel(const float* __restrict__ in,
                                                   bf16* __restrict__ outp, int n) {
    int i = blockIdx.x * 256 + threadIdx.x;
    if (i < n) outp[i] = (bf16)in[i];
}

// ---------------------------------------------------------------------------
// Host orchestration
// ---------------------------------------------------------------------------
#define MB1 (1024ull * 1024ull)

extern "C" void kernel_launch(void* const* d_in, const int* in_sizes, int n_in,
                              void* d_out, int out_size, void* d_ws, size_t ws_size,
                              hipStream_t stream) {
    (void)in_sizes; (void)n_in; (void)out_size; (void)ws_size;

    const float* x          = (const float*)d_in[0];
    const float* tvec       = (const float*)d_in[1];
    const float* norm_scale = (const float*)d_in[2];
    const float* ffn_a_w    = (const float*)d_in[3];
    const float* ffn_a_b    = (const float*)d_in[4];
    const float* ffn_b_w    = (const float*)d_in[5];
    const float* ffn_b_b    = (const float*)d_in[6];
    const float* ffn_c_w    = (const float*)d_in[7];
    const float* ffn_c_b    = (const float*)d_in[8];
    const float* enc_p1_w   = (const float*)d_in[9];
    const float* enc_p1_b   = (const float*)d_in[10];
    const float* enc_p2_w   = (const float*)d_in[11];
    const float* enc_p2_b   = (const float*)d_in[12];
    const float* qkv_w      = (const float*)d_in[13];
    const float* qkv_b      = (const float*)d_in[14];
    const float* attn_out_w = (const float*)d_in[15];
    const float* attn_out_b = (const float*)d_in[16];
    const float* rel_bias   = (const float*)d_in[17];
    float* out = (float*)d_out;

    char* ws = (char*)d_ws;
    // lifetime-overlapped workspace layout
    bf16*  embs  = (bf16*)(ws + 0);            // [T][512] bf16, 64MB   (dead after gemm1)
    bf16*  h1    = (bf16*)(ws + 64 * MB1);     // [T][512] bf16, 64MB   (dead after gemm2)
    bf16*  fa    = (bf16*)(ws + 0);            // [T][1024] bf16, 128MB (reuses embs+h1)
    float* mbuf  = (float*)(ws + 128 * MB1);   // [T][512] f32, 128MB   (dead after LN)
    bf16*  attno = (bf16*)(ws + 128 * MB1);    // [T][256] bf16, 32MB   (reuses mbuf)
    float* attnf = (float*)(ws + 160 * MB1);   // [T][256] f32, 64MB
    float* res   = (float*)(ws + 256 * MB1);   // [T][256] f32, 64MB
    bf16*  xn    = (bf16*)(ws + 320 * MB1);    // [T][256] bf16, 32MB
    bf16*  qkvb  = (bf16*)(ws + 352 * MB1);    // [T][768] bf16, 96MB   (dead after attn)
    bf16*  fb    = (bf16*)(ws + 352 * MB1);    // [T][1024] bf16, 128MB (reuses qkv)
    float* ffno  = (float*)(ws + 480 * MB1);   // [T][256] f32, 64MB
    char*  wbase = ws + 544 * MB1;             // bf16 weights (~3MB)

    bf16* wb_p1 = (bf16*)(wbase);                 // 512*512
    bf16* wb_p2 = wb_p1 + 512 * 512;              // 512*512
    bf16* wb_qk = wb_p2 + 512 * 512;              // 768*256
    bf16* wb_ao = wb_qk + 768 * 256;              // 256*256
    bf16* wb_fa = wb_ao + 256 * 256;              // 1024*256
    bf16* wb_fb = wb_fa + 1024 * 256;             // 1024*256
    bf16* wb_fc = wb_fb + 1024 * 256;             // 256*1024

    auto cvt = [&](const float* src, bf16* dst, int n) {
        f2bf_kernel<<<(n + 255) / 256, 256, 0, stream>>>(src, dst, n);
    };
    cvt(enc_p1_w, wb_p1, 512 * 512);
    cvt(enc_p2_w, wb_p2, 512 * 512);
    cvt(qkv_w,    wb_qk, 768 * 256);
    cvt(attn_out_w, wb_ao, 256 * 256);
    cvt(ffn_a_w,  wb_fa, 1024 * 256);
    cvt(ffn_b_w,  wb_fb, 1024 * 256);
    cvt(ffn_c_w,  wb_fc, 256 * 1024);

    // 1) embeddings
    embs_kernel<<<(T_TOK * 512) / 256, 256, 0, stream>>>(tvec, embs);

    // 2) h1 = relu(embs @ p1^T + b1)   [T,512]x[512,512]
    gemm_wmma_kernel<true, true><<<dim3(T_TOK / 128, 512 / 64), 256, 0, stream>>>(
        embs, wb_p1, enc_p1_b, (void*)h1, 512, 512);

    // 3) mul|bias = h1 @ p2^T + b2     -> f32
    gemm_wmma_kernel<false, false><<<dim3(T_TOK / 128, 512 / 64), 256, 0, stream>>>(
        h1, wb_p2, enc_p2_b, (void*)mbuf, 512, 512);

    // 4) affine + layernorm
    affine_ln_kernel<<<T_TOK / 32, 256, 0, stream>>>(x, mbuf, norm_scale, res, xn);

    // 5) qkv = xn @ qkv_w^T + qkv_b    -> bf16 [T,768]
    gemm_wmma_kernel<false, true><<<dim3(T_TOK / 128, 768 / 64), 256, 0, stream>>>(
        xn, wb_qk, qkv_b, (void*)qkvb, 768, 256);

    // 6) shifted-window attention -> bf16 [T,256]
    attn_kernel<<<4096, 256, 0, stream>>>(qkvb, rel_bias, attno);

    // 7) attn_out projection -> f32
    gemm_wmma_kernel<false, false><<<dim3(T_TOK / 128, 256 / 64), 256, 0, stream>>>(
        attno, wb_ao, attn_out_b, (void*)attnf, 256, 256);

    // 8) FFN branches
    gemm_wmma_kernel<false, true><<<dim3(T_TOK / 128, 1024 / 64), 256, 0, stream>>>(
        xn, wb_fa, ffn_a_b, (void*)fa, 1024, 256);
    gemm_wmma_kernel<true, true><<<dim3(T_TOK / 128, 1024 / 64), 256, 0, stream>>>(
        xn, wb_fb, ffn_b_b, (void*)fb, 1024, 256);
    gate_mul_kernel<<<(T_TOK * 1024) / 256, 256, 0, stream>>>(fa, fb);

    // 9) ffn_c projection -> f32
    gemm_wmma_kernel<false, false><<<dim3(T_TOK / 128, 256 / 64), 256, 0, stream>>>(
        fa, wb_fc, ffn_c_b, (void*)ffno, 256, 1024);

    // 10) out = attn + ffn + res, back to (B,C,H,W)
    final_add_kernel<<<T_TOK / 32, 256, 0, stream>>>(attnf, ffno, res, out);
}
